// Force_37220186587360
// MI455X (gfx1250) — matrix operand order repaired and checked by reading, not compiled, not run on hardware
//
#include <hip/hip_runtime.h>
#include <hip/hip_bf16.h>

#define N_NODES 2048
#define N_EDGES 4096
#define N_MOLS  64
#define V_DIM   128
#define E_DIM   64
#define Q_DIM   3
#define H_DIM   32
#define X_DIM   (V_DIM + E_DIM)   // 192
#define X_PAD   200               // padded LDS row stride (halves) -> bank-conflict free
#define ESPF    1e-6f

typedef __attribute__((ext_vector_type(16))) _Float16 v16h;
typedef __attribute__((ext_vector_type(8)))  _Float16 v8h;
typedef __attribute__((ext_vector_type(8)))  float    v8f;

// ---------------------------------------------------------------------------
// Kernel 0: initialize output + molecule range scratch (deterministic each call)
// ---------------------------------------------------------------------------
__global__ void init_kernel(float* __restrict__ out, int* __restrict__ mol_start,
                            int* __restrict__ mol_end) {
    int stride = gridDim.x * blockDim.x;
    for (int i = blockIdx.x * blockDim.x + threadIdx.x; i < N_NODES * Q_DIM; i += stride)
        out[i] = 0.0f;
    for (int i = blockIdx.x * blockDim.x + threadIdx.x; i < N_MOLS; i += stride) {
        mol_start[i] = N_NODES;
        mol_end[i]   = 0;
    }
}

// ---------------------------------------------------------------------------
// Kernel 1: bandwidth-bound decode of one-hot matrices (float4 streaming)
//   vew1[n,j]!=0 -> u_idx[j]=n ; vew2 -> v_idx ; mvw[mm,n]!=0 -> mol_id[n]=mm
// ---------------------------------------------------------------------------
__global__ void extract_kernel(const float* __restrict__ vew1,
                               const float* __restrict__ vew2,
                               const float* __restrict__ mvw,
                               int* __restrict__ u_idx, int* __restrict__ v_idx,
                               int* __restrict__ mol_id,
                               int* __restrict__ mol_start, int* __restrict__ mol_end) {
    const long stride = (long)gridDim.x * blockDim.x;
    const long tid0   = (long)blockIdx.x * blockDim.x + threadIdx.x;

    const long nvew4 = (long)N_NODES * N_EDGES / 4;
    const float4* vw1 = (const float4*)vew1;
    const float4* vw2 = (const float4*)vew2;
    for (long idx = tid0; idx < nvew4; idx += stride) {
        float4 a = vw1[idx];
        float4 b = vw2[idx];
        long base = idx * 4;
        float av[4] = {a.x, a.y, a.z, a.w};
        float bv[4] = {b.x, b.y, b.z, b.w};
        #pragma unroll
        for (int k = 0; k < 4; ++k) {
            if (av[k] != 0.0f) { long f = base + k; u_idx[f % N_EDGES] = (int)(f / N_EDGES); }
            if (bv[k] != 0.0f) { long f = base + k; v_idx[f % N_EDGES] = (int)(f / N_EDGES); }
        }
    }

    const long nmvw4 = (long)N_MOLS * N_NODES / 4;
    const float4* mw = (const float4*)mvw;
    for (long idx = tid0; idx < nmvw4; idx += stride) {
        float4 c = mw[idx];
        long base = idx * 4;
        float cv[4] = {c.x, c.y, c.z, c.w};
        #pragma unroll
        for (int k = 0; k < 4; ++k) {
            if (cv[k] != 0.0f) {
                long f = base + k;
                int mm = (int)(f / N_NODES);
                int n  = (int)(f % N_NODES);
                mol_id[n] = mm;
                atomicMin(&mol_start[mm], n);
                atomicMax(&mol_end[mm], n + 1);
            }
        }
    }
}

// ---------------------------------------------------------------------------
// Kernel 2: bond branch with WMMA.
//   Rows j in [0, 2E). Row j: x = [ v[u]+v[vv] (128) , e[j%E] (64) ]  (192)
//   h = leaky_relu(x @ W_fb1) [32] ; value = tanh(h @ W_fb2 + b_fb2)
//   scatter: out[u] += (q[u]-q[vv]) / (|dq|+eps) * value
// 4 waves/block, each wave owns a 16-row tile -> 12 v_wmma_f32_16x16x32_f16.
// W stored TRANSPOSED ([col][K]) + padded rows so every fragment operand is
// two aligned ds_load_b128 per lane, no bank conflicts.
// ---------------------------------------------------------------------------
__global__ __launch_bounds__(128) void bond_kernel(
        const float* __restrict__ v, const float* __restrict__ e,
        const float* __restrict__ q,
        const float* __restrict__ W1, const float* __restrict__ W2,
        const float* __restrict__ b2,
        const int* __restrict__ u_idx, const int* __restrict__ v_idx,
        float* __restrict__ out) {
    __shared__ _Float16 sWT[H_DIM * X_PAD];       // W_fb1^T: [32 cols][200] f16
    __shared__ _Float16 sX[4][16 * X_PAD];        // 4 waves x 16 rows x 200 f16
    __shared__ float    sVal[4][16];

    const int tid  = threadIdx.x;
    const int wave = tid >> 5;
    const int lane = tid & 31;
    const int rsel = lane & 15;
    const int hi   = lane >> 4;

    // W_fb1 [192][32] -> transposed f16 LDS [32][200] (shared by all waves)
    for (int t = tid; t < X_DIM * H_DIM; t += 128) {
        int k = t >> 5;            // K index   (t / 32)
        int col = t & 31;          // column    (t % 32)
        sWT[col * X_PAD + k] = (_Float16)W1[t];
    }

    // Stage this wave's 16 input rows into LDS as f16
    const int rowbase = blockIdx.x * 64 + wave * 16;
    for (int r = 0; r < 16; ++r) {
        int j    = rowbase + r;
        int u    = (j < N_EDGES) ? u_idx[j] : v_idx[j - N_EDGES];
        int vv   = (j < N_EDGES) ? v_idx[j] : u_idx[j - N_EDGES];
        int eidx = (j < N_EDGES) ? j : j - N_EDGES;
        for (int t = lane; t < X_DIM; t += 32) {  // uniform: 6 iters for every lane
            float val = (t < V_DIM) ? (v[u * V_DIM + t] + v[vv * V_DIM + t])
                                    : e[eidx * E_DIM + (t - V_DIM)];
            sX[wave][r * X_PAD + t] = (_Float16)val;
        }
    }
    __syncthreads();

    // GEMM: 16x192 (A) x 192x32 (B) -> 16x32, two 16-col accumulators
    const _Float16* xrow  = &sX[wave][rsel * X_PAD];
    const _Float16* wcol0 = &sWT[rsel * X_PAD];
    const _Float16* wcol1 = &sWT[(rsel + 16) * X_PAD];

    v8f acc0 = {}; v8f acc1 = {};
    #pragma unroll
    for (int c = 0; c < X_DIM / 32; ++c) {        // 6 K-chunks of 32
        const int kb = c * 32;
        // A fragment (contiguous k order; B uses the same):
        //   a[0..15] = 16 contiguous K at kb + hi*16 of this lane's row
        v8h a_lo = *(const v8h*)&xrow[kb + hi * 16];
        v8h a_hi = *(const v8h*)&xrow[kb + hi * 16 + 8];
        // B fragments (lane = column, hi half -> K+16; contiguous k order, same as A):
        //   b[0..15] = 16 contiguous K at kb + hi*16 of this lane's column
        v8h b0_lo = *(const v8h*)&wcol0[kb + hi * 16];
        v8h b0_hi = *(const v8h*)&wcol0[kb + hi * 16 + 8];
        v8h b1_lo = *(const v8h*)&wcol1[kb + hi * 16];
        v8h b1_hi = *(const v8h*)&wcol1[kb + hi * 16 + 8];

        v16h a, b0, b1;
        #pragma unroll
        for (int i = 0; i < 8; ++i) {
            a[i]  = a_lo[i];  a[i + 8]  = a_hi[i];
            b0[i] = b0_lo[i]; b0[i + 8] = b0_hi[i];
            b1[i] = b1_lo[i]; b1[i + 8] = b1_hi[i];
        }
        acc0 = __builtin_amdgcn_wmma_f32_16x16x32_f16(false, a, false, b0,
                                                      (short)0, acc0, false, false);
        acc1 = __builtin_amdgcn_wmma_f32_16x16x32_f16(false, a, false, b1,
                                                      (short)0, acc1, false, false);
    }

    // leaky_relu + fused dot with W_fb2 (col = rsel for acc0, rsel+16 for acc1)
    float w2a = W2[rsel], w2b = W2[rsel + 16];
    float partial[8];
    #pragma unroll
    for (int i = 0; i < 8; ++i) {
        float h0 = acc0[i]; h0 = (h0 > 0.0f) ? h0 : 0.01f * h0;
        float h1 = acc1[i]; h1 = (h1 > 0.0f) ? h1 : 0.01f * h1;
        partial[i] = h0 * w2a + h1 * w2b;
    }
    // butterfly-reduce within each 16-lane group (rows 0-7 in group0, 8-15 in group1)
    #pragma unroll
    for (int msk = 1; msk < 16; msk <<= 1) {
        #pragma unroll
        for (int i = 0; i < 8; ++i)
            partial[i] += __shfl_xor(partial[i], msk, 32);
    }
    if (rsel == 0) {
        #pragma unroll
        for (int i = 0; i < 8; ++i)
            sVal[wave][hi * 8 + i] = partial[i];
    }
    __syncthreads();

    // one lane per row: tanh, unit vector, scatter-add
    if (lane < 16) {
        int j  = rowbase + lane;
        int u  = (j < N_EDGES) ? u_idx[j] : v_idx[j - N_EDGES];
        int vv = (j < N_EDGES) ? v_idx[j] : u_idx[j - N_EDGES];
        float value = tanhf(sVal[wave][lane] + b2[0]);
        float d0 = q[u * 3 + 0] - q[vv * 3 + 0];
        float d1 = q[u * 3 + 1] - q[vv * 3 + 1];
        float d2 = q[u * 3 + 2] - q[vv * 3 + 2];
        float s  = value / (sqrtf(d0 * d0 + d1 * d1 + d2 * d2) + ESPF);
        atomicAdd(&out[u * 3 + 0], d0 * s);
        atomicAdd(&out[u * 3 + 1], d1 * s);
        atomicAdd(&out[u * 3 + 2], d2 * s);
    }
}

// ---------------------------------------------------------------------------
// Kernel 3: pairwise branch, exact molecule pruning (mass_mask==0 off-molecule
// and tanh(0)*unit == 0 exactly). One wave per node i; lanes split the
// contiguous same-molecule range (mol_id sorted).
// ---------------------------------------------------------------------------
__global__ __launch_bounds__(256) void pair_kernel(
        const float* __restrict__ q, const float* __restrict__ m,
        const int* __restrict__ mol_id,
        const int* __restrict__ mol_start, const int* __restrict__ mol_end,
        const float* __restrict__ W_fr, const float* __restrict__ b_fr,
        float* __restrict__ out) {
    __shared__ float sW[Q_DIM * H_DIM];   // 96
    __shared__ float sb[H_DIM];           // 32
    const int tid = threadIdx.x;
    if (tid < Q_DIM * H_DIM) sW[tid] = W_fr[tid];
    if (tid < H_DIM)         sb[tid] = b_fr[tid];
    __syncthreads();

    const int wave = tid >> 5;
    const int lane = tid & 31;
    const int i    = blockIdx.x * 8 + wave;
    if (i >= N_NODES) return;

    const int mm = mol_id[i];
    const int lo = mol_start[mm];
    const int hiE = mol_end[mm];
    const float qi0 = q[i * 3 + 0], qi1 = q[i * 3 + 1], qi2 = q[i * 3 + 2];
    const float mi = m[i];

    float a0 = 0.0f, a1 = 0.0f, a2 = 0.0f;
    for (int j = lo + lane; j < hiE; j += 32) {
        float d0 = qi0 - q[j * 3 + 0];
        float d1 = qi1 - q[j * 3 + 1];
        float d2 = qi2 - q[j * 3 + 2];
        float ss = 0.0f;
        #pragma unroll
        for (int k = 0; k < H_DIM; ++k) {
            float t = d0 * sW[k] + d1 * sW[H_DIM + k] + d2 * sW[2 * H_DIM + k] + sb[k];
            // numerically stable softplus
            float sp = (t > 0.0f) ? (t + log1pf(expf(-t))) : log1pf(expf(t));
            ss += sp * sp;
        }
        float delta = sqrtf(ss);
        float val = tanhf((1.0f / (delta * delta) - 1.0f / delta) * (mi * m[j]));
        float rn  = sqrtf(d0 * d0 + d1 * d1 + d2 * d2);
        float inv = val / (rn + ESPF);     // j==i -> d==0 -> exact zero contribution
        a0 += d0 * inv; a1 += d1 * inv; a2 += d2 * inv;
    }
    #pragma unroll
    for (int msk = 1; msk < 32; msk <<= 1) {
        a0 += __shfl_xor(a0, msk, 32);
        a1 += __shfl_xor(a1, msk, 32);
        a2 += __shfl_xor(a2, msk, 32);
    }
    if (lane == 0) {   // stream-ordered after bond_kernel: plain RMW is safe
        out[i * 3 + 0] += a0;
        out[i * 3 + 1] += a1;
        out[i * 3 + 2] += a2;
    }
}

// ---------------------------------------------------------------------------
extern "C" void kernel_launch(void* const* d_in, const int* in_sizes, int n_in,
                              void* d_out, int out_size, void* d_ws, size_t ws_size,
                              hipStream_t stream) {
    const float* v    = (const float*)d_in[0];
    const float* e    = (const float*)d_in[1];
    const float* m    = (const float*)d_in[2];
    const float* q    = (const float*)d_in[3];
    const float* vew1 = (const float*)d_in[4];
    const float* vew2 = (const float*)d_in[5];
    const float* mvw  = (const float*)d_in[6];
    const float* W1   = (const float*)d_in[7];
    const float* W2   = (const float*)d_in[8];
    const float* b2   = (const float*)d_in[9];
    const float* Wfr  = (const float*)d_in[10];
    const float* bfr  = (const float*)d_in[11];
    float* out = (float*)d_out;

    int* u_idx     = (int*)d_ws;
    int* v_idx     = u_idx + N_EDGES;
    int* mol_id    = v_idx + N_EDGES;
    int* mol_start = mol_id + N_NODES;
    int* mol_end   = mol_start + N_MOLS;

    init_kernel<<<dim3(32), 256, 0, stream>>>(out, mol_start, mol_end);
    extract_kernel<<<dim3(2048), 256, 0, stream>>>(vew1, vew2, mvw,
                                                   u_idx, v_idx, mol_id,
                                                   mol_start, mol_end);
    bond_kernel<<<dim3((2 * N_EDGES) / 64), 128, 0, stream>>>(v, e, q, W1, W2, b2,
                                                              u_idx, v_idx, out);
    pair_kernel<<<dim3(N_NODES / 8), 256, 0, stream>>>(q, m, mol_id, mol_start,
                                                       mol_end, Wfr, bfr, out);
}
